// LA_TPP_Model_48979807044206
// MI455X (gfx1250) — matrix operand order repaired and checked by reading, not compiled, not run on hardware
//
#include <hip/hip_runtime.h>

typedef __attribute__((ext_vector_type(16))) __bf16 v16bf;
typedef __attribute__((ext_vector_type(8)))  float  v8f;

#define NPRED 256
#define RDIM  256
#define MDIM  64
#define ADIM  64
#define EDIM  32
#define OUTW  512
#define ROWS_PER_WG 128
// bf16-element offsets inside d_ws for the three fragment-swizzled B matrices
#define GM_WS 0
#define GA_WS 16384
#define W_WS  32768
#define WS_TOTAL 36864   // bf16 elements

union FragB {
    v16bf v;
    uint4 q[2];
    __bf16 h[16];
};

// Pack one 16-bit A-matrix fragment (16x32 tile, this lane's 16 values) from
// f32 global memory using native f32->bf16 hardware conversion.
// Per ISA layout: lane half `hi` owns K = kb+8*hi+[0,8) (elements 0..7) and
// K = kb+16+8*hi+[0,8) (elements 8..15).
__device__ __forceinline__ void packA(FragB& f, const float* __restrict__ src) {
    float4 c0 = *(const float4*)(src);
    float4 c1 = *(const float4*)(src + 4);
    float4 c2 = *(const float4*)(src + 16);
    float4 c3 = *(const float4*)(src + 20);
    f.h[0]  = (__bf16)c0.x; f.h[1]  = (__bf16)c0.y; f.h[2]  = (__bf16)c0.z; f.h[3]  = (__bf16)c0.w;
    f.h[4]  = (__bf16)c1.x; f.h[5]  = (__bf16)c1.y; f.h[6]  = (__bf16)c1.z; f.h[7]  = (__bf16)c1.w;
    f.h[8]  = (__bf16)c2.x; f.h[9]  = (__bf16)c2.y; f.h[10] = (__bf16)c2.z; f.h[11] = (__bf16)c2.w;
    f.h[12] = (__bf16)c3.x; f.h[13] = (__bf16)c3.y; f.h[14] = (__bf16)c3.z; f.h[15] = (__bf16)c3.w;
}

// ---------------------------------------------------------------------------
// Kernel 1: fold masks into gamma matrices, convert to bf16, and store in
// WMMA B-fragment order so the compute waves can load fragments with two
// coalesced b128 reads per lane. B-fragment layout (16-bit B, 32x16):
//   element e -> K = kk*32 + (e&7) + 8*(lane>=16) + 16*(e>>3), N = nt*16 + (lane&15)
// ---------------------------------------------------------------------------
__global__ __launch_bounds__(256) void swizzle_kernel(
    const float* __restrict__ gAM, const float* __restrict__ gMA,
    const float* __restrict__ w,
    const unsigned char* __restrict__ maskM, const unsigned char* __restrict__ maskA,
    __bf16* __restrict__ ws)
{
    int i = blockIdx.x * 256 + threadIdx.x;
    if (i >= WS_TOTAL) return;
    float val;
    int dest;
    if (i < 32768) {
        // gamma_AtoM / gamma_MtoA : [R=256, 64], B[k][n] = gamma[k*64+n]*mask[k]
        bool isGM = i < 16384;
        int f = i & 16383;
        int K = f >> 6, N = f & 63;
        int kk = K >> 5, r = K & 31, hi = (r >> 3) & 1, e = (r & 7) + ((r >> 4) << 3);
        int lane = (N & 15) + (hi << 4), nt = N >> 4;
        const float* src = isGM ? gAM : gMA;
        const unsigned char* msk = isGM ? maskM : maskA;
        val = src[f] * (msk[K] ? 1.0f : 0.0f);
        dest = (isGM ? GM_WS : GA_WS) + (kk * 4 + nt) * 512 + lane * 16 + e;
    } else {
        // w : [a=64, m=64], GEMM2 B[k][n] = w[n*64+k]  (flat = n*64+k)
        int f = i - 32768;
        int n = f >> 6, k = f & 63;
        int kk = k >> 5, r = k & 31, hi = (r >> 3) & 1, e = (r & 7) + ((r >> 4) << 3);
        int lane = (n & 15) + (hi << 4), nt = n >> 4;
        val = w[f];
        dest = W_WS + (kk * 4 + nt) * 512 + lane * 16 + e;
    }
    ws[dest] = (__bf16)val;
}

// ---------------------------------------------------------------------------
// Kernel 2: fully fused. 8 wave32s per block; each wave owns a 16-row tile.
// ---------------------------------------------------------------------------
__global__ __launch_bounds__(256) void fused_kernel(
    const float* __restrict__ M, const float* __restrict__ g,
    const float* __restrict__ bvec,
    const float* __restrict__ hist_ts, const float* __restrict__ t_now,
    const int* __restrict__ hist_pos, const int* __restrict__ mental_pos,
    const __bf16* __restrict__ ws, float* __restrict__ out, int B)
{
    __shared__ float rowbuf[ROWS_PER_WG * 128];   // 64 KB: val positions [0,128)
    const int tid = threadIdx.x;
    const long rowBase = (long)blockIdx.x * ROWS_PER_WG;

    // ---- val phase ----------------------------------------------------------
    #pragma unroll 4
    for (int i = tid; i < ROWS_PER_WG * 128; i += 256) rowbuf[i] = 0.f;
    __syncthreads();
    // at.set: scatter M into mental positions (setup guarantees pos < m=64)
    for (int i = tid; i < ROWS_PER_WG * MDIM; i += 256) {
        int r = i >> 6, j = i & 63;
        long row = rowBase + r;
        if (row < B) {
            int pos = mental_pos[j];
            if ((unsigned)pos < 128u) rowbuf[r * 128 + pos] = M[row * MDIM + j];
        }
    }
    __syncthreads();
    // at.max of decay-gated writes; ok mask restricts to action preds [64,128)
    if (tid < ROWS_PER_WG) {
        long row = rowBase + tid;
        if (row < B) {
            float tn = t_now[row];
            #pragma unroll
            for (int e = 0; e < EDIM; ++e) {
                float ts = hist_ts[row * EDIM + e];
                int pos  = hist_pos[row * EDIM + e];
                float td = tn - ts;
                if (pos >= MDIM && pos < MDIM + ADIM && td <= 1.0f + 1e-9f) {
                    float dec = fmaxf(0.1f, 1.0f - td * 0.5f);
                    float* p = &rowbuf[tid * 128 + pos];
                    *p = fmaxf(*p, dec);
                }
            }
        }
    }
    __syncthreads();
    // coalesced write of val[0:256] (tail [128,256) is zero)
    for (int i = tid; i < ROWS_PER_WG * NPRED; i += 256) {
        int r = i >> 8, p = i & 255;
        long row = rowBase + r;
        if (row < B) out[row * OUTW + p] = (p < 128) ? rowbuf[r * 128 + p] : 0.f;
    }

    // ---- GEMM phase (per wave32, one 16-row tile) ---------------------------
    const int lane = tid & 31;
    const int wave = tid >> 5;
    const int lrow = lane & 15;
    const int hi   = lane >> 4;
    const long rt  = rowBase + (long)wave * 16;
    long arow = rt + lrow;
    if (arow >= B) arow = B - 1;   // clamp (tiles are full for B % 128 == 0)

    // A fragments of g (reused for both gamma GEMMs): 8 k-steps of 32
    FragB ag[8];
    #pragma unroll
    for (int kk = 0; kk < 8; ++kk)
        packA(ag[kk], g + arow * RDIM + kk * 32 + hi * 8);

    v8f accS[4], accL[4], accZ[4];
    #pragma unroll
    for (int nt = 0; nt < 4; ++nt) {
        v8f z = 0.0f;
        accS[nt] = z; accL[nt] = z; accZ[nt] = z;
    }

    const uint4* wsq = (const uint4*)ws;  // 8 bf16 per uint4
    #pragma unroll
    for (int kk = 0; kk < 8; ++kk) {
        #pragma unroll
        for (int nt = 0; nt < 4; ++nt) {
            const int fb = (((kk * 4 + nt) * 512) + lane * 16) >> 3;
            FragB bm; bm.q[0] = wsq[(GM_WS >> 3) + fb]; bm.q[1] = wsq[(GM_WS >> 3) + fb + 1];
            accS[nt] = __builtin_amdgcn_wmma_f32_16x16x32_bf16(
                false, ag[kk].v, false, bm.v, (short)0, accS[nt], false, false);
            FragB ba; ba.q[0] = wsq[(GA_WS >> 3) + fb]; ba.q[1] = wsq[(GA_WS >> 3) + fb + 1];
            accL[nt] = __builtin_amdgcn_wmma_f32_16x16x32_bf16(
                false, ag[kk].v, false, ba.v, (short)0, accL[nt], false, false);
        }
    }

    // GEMM2: z = M @ w^T  (K = m = 64 -> 2 k-steps)
    FragB am[2];
    #pragma unroll
    for (int kk = 0; kk < 2; ++kk)
        packA(am[kk], M + arow * MDIM + kk * 32 + hi * 8);
    #pragma unroll
    for (int kk = 0; kk < 2; ++kk) {
        #pragma unroll
        for (int nt = 0; nt < 4; ++nt) {
            const int fb = (((kk * 4 + nt) * 512) + lane * 16) >> 3;
            FragB bw; bw.q[0] = wsq[(W_WS >> 3) + fb]; bw.q[1] = wsq[(W_WS >> 3) + fb + 1];
            accZ[nt] = __builtin_amdgcn_wmma_f32_16x16x32_bf16(
                false, am[kk].v, false, bw.v, (short)0, accZ[nt], false, false);
        }
    }

    // Epilogue: relu / softplus + stores. D layout: VGPR v -> row rt+v+8*hi,
    // col = nt*16 + (lane&15)  (lanes 0-15 write 64B-contiguous runs)
    #pragma unroll
    for (int nt = 0; nt < 4; ++nt) {
        const int col = nt * 16 + lrow;
        const float bc = bvec[col];
        #pragma unroll
        for (int v = 0; v < 8; ++v) {
            long row = rt + v + hi * 8;
            if (row < B) {
                float s  = fmaxf(accS[nt][v], 0.f);
                float ll = fmaxf(accL[nt][v], 0.f);
                float x  = bc + accZ[nt][v];
                float lm = fmaxf(x, 0.f) + log1pf(expf(-fabsf(x)));  // stable softplus
                float* o = out + row * OUTW;
                o[256 + col] = s;        // S
                o[320 + col] = lm + ll;  // total_intensity
                o[384 + col] = lm;       // lam_ment
                o[448 + col] = ll;       // lam_logic
            }
        }
    }
}

extern "C" void kernel_launch(void* const* d_in, const int* in_sizes, int n_in,
                              void* d_out, int out_size, void* d_ws, size_t ws_size,
                              hipStream_t stream) {
    (void)n_in; (void)out_size; (void)ws_size;
    const float* M        = (const float*)d_in[0];
    const float* g        = (const float*)d_in[1];
    const float* gAM      = (const float*)d_in[2];
    const float* gMA      = (const float*)d_in[3];
    const float* b        = (const float*)d_in[4];
    const float* w        = (const float*)d_in[5];
    const float* hist_ts  = (const float*)d_in[6];
    const float* t_now    = (const float*)d_in[7];
    const int*   hist_pos = (const int*)d_in[8];
    const int*   ment_pos = (const int*)d_in[9];
    const unsigned char* maskM = (const unsigned char*)d_in[10];
    const unsigned char* maskA = (const unsigned char*)d_in[11];
    float* out = (float*)d_out;
    __bf16* ws = (__bf16*)d_ws;

    const int B = in_sizes[0] / MDIM;

    swizzle_kernel<<<(WS_TOTAL + 255) / 256, 256, 0, stream>>>(gAM, gMA, w, maskM, maskA, ws);

    const int nblocks = (B + ROWS_PER_WG - 1) / ROWS_PER_WG;
    fused_kernel<<<nblocks, 256, 0, stream>>>(M, g, b, hist_ts, t_now,
                                              hist_pos, ment_pos, ws, out, B);
}
